// HardNegativeMiner_47124381172078
// MI455X (gfx1250) — matrix operand order repaired and checked by reading, not compiled, not run on hardware
//
#include <hip/hip_runtime.h>
#include <hip/hip_bf16.h>

// ---------------------------------------------------------------------------
// HardNegativeMiner for MI455X (gfx1250, wave32, WMMA)
//   K1: row L2-normalize memory  f32 -> f16   (ws)
//   K2: fused  sim^T = mem_tile @ src_n^T  +  mask  +  per-row top-5.
//       v_wmma_f32_16x16x32_f16, src fragments resident in 64 VGPRs,
//       TWO independent accumulator chains per iteration (2 column tiles)
//       so a single wave keeps the matrix pipe busy; incremental 64-bit
//       pointers (no per-tile clamp math) + global_prefetch one step ahead.
//   K3: stable partition + deterministic random fill + smoothing + assembly.
// ---------------------------------------------------------------------------

typedef __attribute__((ext_vector_type(16))) _Float16 v16h;
typedef __attribute__((ext_vector_type(8)))  float    v8f;

#define DIM      256
#define TOPK     5
#define FILLV    (-1.0e9f)
#define REAL_TH  (-1.0e9f + 1.0f)   // is_real = score > FILL + 1
#define NEGINF   (-3.0e38f)

// ---------------- K1: normalize rows of memory, emit f16 -------------------
__global__ void hnm_normalize_kernel(const float* __restrict__ mem,
                                     _Float16* __restrict__ memH) {
    __shared__ float red[DIM];
    const int row = blockIdx.x;
    const int t   = threadIdx.x;           // blockDim.x == 256 == DIM
    const float x = mem[(size_t)row * DIM + t];
    red[t] = x * x;
    __syncthreads();
    #pragma unroll
    for (int s = DIM / 2; s > 0; s >>= 1) {
        if (t < s) red[t] += red[t + s];
        __syncthreads();
    }
    const float nrm   = sqrtf(red[0]);
    const float scale = 1.0f / fmaxf(nrm, 1e-12f);
    memH[(size_t)row * DIM + t] = (_Float16)(x * scale);
}

// sorted-descending top-5 insert (strict > keeps earliest on ties)
__device__ __forceinline__ void hnm_topk_insert(float (&tv)[TOPK], int (&ti)[TOPK],
                                                float score, int id) {
    if (score > tv[TOPK - 1]) {
        float v = score; int idv = id;
        #pragma unroll
        for (int p = 0; p < TOPK; ++p) {
            if (v > tv[p]) {
                const float a = tv[p]; const int b = ti[p];
                tv[p] = v; ti[p] = idv; v = a; idv = b;
            }
        }
    }
}

// generic (clamped, bounds-checked) single 16-column tile — tail only
__device__ __forceinline__ void hnm_tile_single(const _Float16* __restrict__ memH,
                                                const v16h (&bfr)[8],
                                                int col0, int grp, int half, int N,
                                                int dstMy, bool labOK,
                                                float (&tv)[TOPK], int (&ti)[TOPK]) {
    const int memRow = min(col0 + grp, N - 1);
    const _Float16* ap = memH + (size_t)memRow * DIM + half * 8;
    v8f acc = {};
    #pragma unroll
    for (int kc = 0; kc < 8; ++kc) {
        const _Float16* p = ap + kc * 32;
        v16h a;
        #pragma unroll
        for (int j = 0; j < 8; ++j) { a[j] = p[j]; a[j + 8] = p[j + 16]; }
        acc = __builtin_amdgcn_wmma_f32_16x16x32_f16(
                  false, a, false, bfr[kc], (short)0, acc, false, false);
    }
    #pragma unroll
    for (int i = 0; i < 8; ++i) {
        const int col = col0 + half * 8 + i;
        const bool valid = (col < N) && (col != dstMy) && (acc[i] > 0.1f) && labOK;
        hnm_topk_insert(tv, ti, (valid ? acc[i] : FILLV) + 1.0f, col);
    }
}

// ---------------- K2: fused GEMM^T + mask + top-5 ---------------------------
// grid.x = B/16 workgroups; block = 256 threads = 8 waves.
// Wave w owns column tiles {2w, 2w+1} (mod 16): two tiles per iteration.
__global__ __launch_bounds__(256, 1)
void hnm_mine_kernel(const _Float16* __restrict__ memH,
                     const int*   __restrict__ src,
                     const int*   __restrict__ dst,
                     const float* __restrict__ labels,
                     float* __restrict__ topkVals,  // [B][5] (into d_out)
                     int*   __restrict__ topkIdx,   // [B][5] (ws)
                     int N) {
    __shared__ float candV[16 * 8 * TOPK];
    __shared__ int   candI[16 * 8 * TOPK];
    __shared__ int   srcS[16];
    __shared__ int   dstS[16];
    __shared__ float labS[16];

    const int rowBase = blockIdx.x * 16;
    const int t = threadIdx.x;
    if (t < 16) {
        srcS[t] = src[rowBase + t];
        dstS[t] = dst[rowBase + t];
        labS[t] = labels[rowBase + t];
    }
    __syncthreads();

    const int wave = t >> 5;
    const int lane = t & 31;
    const int grp  = lane & 15;   // this lane's src row (C/D layout: N = lane%16)
    const int half = lane >> 4;   // K-subrange / column-subrange selector

    const int  dstMy  = dstS[grp];
    const bool labOK  = (labS[grp] == 1.0f);
    const int  srcIdx = srcS[grp];

    // Loop-invariant B fragments: src_n row, per 32x16 f16 B layout
    //   lane<16 holds K = kc*32 + {0..7, 16..23} ; lane>=16 holds K = kc*32 + {8..15, 24..31}
    v16h bfr[8];
    {
        const _Float16* sp = memH + (size_t)srcIdx * DIM + half * 8;
        #pragma unroll
        for (int kc = 0; kc < 8; ++kc) {
            #pragma unroll
            for (int j = 0; j < 8; ++j) { bfr[kc][j] = sp[kc * 32 + j]; bfr[kc][j + 8] = sp[kc * 32 + 16 + j]; }
        }
    }

    float tv[TOPK]; int ti[TOPK];
    #pragma unroll
    for (int p = 0; p < TOPK; ++p) { tv[p] = NEGINF; ti[p] = 0; }

    const int nTilesFull = N >> 4;            // tiles fully in range (no clamp)
    const int nTiles     = (N + 15) >> 4;
    const size_t STEPH   = (size_t)256 * DIM; // halves per 16-tile stride

    int tile  = wave * 2;
    int colA0 = tile * 16;
    // A fragment pointers for the lane's row of each tile (16x32 f16 A layout:
    //   lane<16: K {kc*32+0..7, +16..23}; lane>=16: {+8..15, +24..31})
    const _Float16* pA = memH + ((size_t)colA0 + grp) * DIM + half * 8;
    const _Float16* pB = pA + (size_t)16 * DIM;

    for (; tile + 2 <= nTilesFull; tile += 16) {
        __builtin_prefetch((const void*)(pA + STEPH), 0, 1);
        __builtin_prefetch((const void*)(pB + STEPH), 0, 1);

        v8f accA = {};
        v8f accB = {};
        #pragma unroll
        for (int kc = 0; kc < 8; ++kc) {
            const _Float16* qa = pA + kc * 32;
            const _Float16* qb = pB + kc * 32;
            v16h a0, a1;
            #pragma unroll
            for (int j = 0; j < 8; ++j) {
                a0[j] = qa[j]; a0[j + 8] = qa[j + 16];
                a1[j] = qb[j]; a1[j + 8] = qb[j + 16];
            }
            accA = __builtin_amdgcn_wmma_f32_16x16x32_f16(
                       false, a0, false, bfr[kc], (short)0, accA, false, false);
            accB = __builtin_amdgcn_wmma_f32_16x16x32_f16(
                       false, a1, false, bfr[kc], (short)0, accB, false, false);
        }

        #pragma unroll
        for (int i = 0; i < 8; ++i) {
            const int colA = colA0 + half * 8 + i;
            const bool vA = (colA != dstMy) && (accA[i] > 0.1f) && labOK;
            hnm_topk_insert(tv, ti, (vA ? accA[i] : FILLV) + 1.0f, colA);

            const int colB = colA + 16;
            const bool vB = (colB != dstMy) && (accB[i] > 0.1f) && labOK;
            hnm_topk_insert(tv, ti, (vB ? accB[i] : FILLV) + 1.0f, colB);
        }

        pA += STEPH; pB += STEPH; colA0 += 256;
    }

    // generic tail (never taken for N % 16 == 0 with even tile pairs)
    for (int tl = tile; tl < nTiles && tl < tile + 2; ++tl)
        hnm_tile_single(memH, bfr, tl * 16, grp, half, N, dstMy, labOK, tv, ti);

    // Merge with partner lane (same row, other 8-column subset)
    {
        float ov[TOPK]; int oi[TOPK];
        #pragma unroll
        for (int p = 0; p < TOPK; ++p) {
            ov[p] = __shfl_xor(tv[p], 16);
            oi[p] = __shfl_xor(ti[p], 16);
        }
        #pragma unroll
        for (int p = 0; p < TOPK; ++p)
            hnm_topk_insert(tv, ti, ov[p], oi[p]);
    }

    // per-wave lists -> LDS
    if (lane < 16) {
        const int base = (grp * 8 + wave) * TOPK;
        #pragma unroll
        for (int p = 0; p < TOPK; ++p) { candV[base + p] = tv[p]; candI[base + p] = ti[p]; }
    }
    __syncthreads();

    // final merge across the 8 waves: one thread per output row
    if (t < 16) {
        float bv[TOPK]; int bi[TOPK];
        #pragma unroll
        for (int p = 0; p < TOPK; ++p) { bv[p] = NEGINF; bi[p] = 0; }
        for (int w = 0; w < 8; ++w) {
            const int base = (t * 8 + w) * TOPK;
            #pragma unroll
            for (int p = 0; p < TOPK; ++p)
                hnm_topk_insert(bv, bi, candV[base + p], candI[base + p]);
        }
        const int gRow = rowBase + t;
        #pragma unroll
        for (int p = 0; p < TOPK; ++p) {
            topkVals[gRow * TOPK + p] = bv[p];
            topkIdx [gRow * TOPK + p] = bi[p];
        }
    }
}

// ---------------- K3: stable partition + fill + assembly --------------------
__device__ __forceinline__ unsigned hnm_hash(unsigned x) {
    x ^= 0x2545F491u;                 // deterministic stand-in for threefry(key=42)
    x *= 0x9E3779B1u; x ^= x >> 16;
    x *= 0x85EBCA6Bu; x ^= x >> 13;
    x *= 0xC2B2AE35u; x ^= x >> 16;
    return x;
}

__global__ void hnm_assemble_kernel(const int*   __restrict__ src,
                                    const int*   __restrict__ dst,
                                    const float* __restrict__ labels,
                                    const float* __restrict__ ts,
                                    const int*   __restrict__ topkIdx,
                                    float* __restrict__ out,
                                    int B, int N) {
    const int M = B * TOPK;          // 20480
    const int T = B + M;             // 24576
    const float* topkVals = out + 4 * (size_t)T;

    __shared__ int scanS[256];
    __shared__ int curS;

    const int t    = threadIdx.x;    // single block of 256
    const int per  = (M + 255) / 256;
    const int begn = t * per;
    const int endn = min(begn + per, M);

    int cnt = 0;
    for (int j = begn; j < endn; ++j) cnt += (topkVals[j] > REAL_TH) ? 1 : 0;
    scanS[t] = cnt;
    __syncthreads();
    if (t == 0) {                    // tiny serial exclusive scan (256 entries)
        int run = 0;
        for (int i = 0; i < 256; ++i) { const int c = scanS[i]; scanS[i] = run; run += c; }
        curS = run;
    }
    __syncthreads();
    const int cur = curS;

    // scatter real negatives to stable ranks [0, cur)
    int rank = scanS[t];
    for (int j = begn; j < endn; ++j) {
        if (topkVals[j] > REAL_TH) {
            const int d = rank++;
            const int b = j / TOPK;
            out[0 * (size_t)T + B + d] = (float)src[b];
            out[1 * (size_t)T + B + d] = (float)topkIdx[j];
            out[3 * (size_t)T + B + d] = ts[b];
        }
    }
    // random fill for [cur, M)  (disjoint from scatter range; no barrier needed)
    for (int j = begn; j < endn; ++j) {
        if (j >= cur) {
            const int fi = j - cur;               // clip(pos-cur, 0, M-1)
            const int b  = fi / TOPK;
            const int rs = src[b];
            int rd = (int)(hnm_hash((unsigned)fi) % (unsigned)N);
            if (rd == rs) rd = (rd + 1) % N;      // collision fix
            out[0 * (size_t)T + B + j] = (float)rs;
            out[1 * (size_t)T + B + j] = (float)rd;
            out[3 * (size_t)T + B + j] = ts[b];
        }
        out[2 * (size_t)T + B + j] = 0.1f;        // smoothed negative label
    }
    // head: original batch + smoothed labels
    for (int i = t; i < B; i += 256) {
        const float lb = labels[i];
        out[0 * (size_t)T + i] = (float)src[i];
        out[1 * (size_t)T + i] = (float)dst[i];
        out[2 * (size_t)T + i] = lb * 0.9f + (1.0f - lb) * 0.1f;
        out[3 * (size_t)T + i] = ts[i];
    }
}

// ---------------------------------------------------------------------------
extern "C" void kernel_launch(void* const* d_in, const int* in_sizes, int n_in,
                              void* d_out, int out_size, void* d_ws, size_t ws_size,
                              hipStream_t stream) {
    const int*   src    = (const int*)  d_in[0];
    const int*   dst    = (const int*)  d_in[1];
    const float* labels = (const float*)d_in[2];
    const float* ts     = (const float*)d_in[3];
    const float* memory = (const float*)d_in[4];

    const int B = in_sizes[0];                    // 4096
    const int N = in_sizes[4] / DIM;              // 100000
    const int M = B * TOPK;                       // 20480
    const int T = B + M;                          // 24576

    float* out = (float*)d_out;

    // workspace layout: [ mem_n f16 : N*256*2 B ][ topk indices : M*4 B ]
    _Float16* memH = (_Float16*)d_ws;
    int* topkIdx   = (int*)((char*)d_ws + (size_t)N * DIM * sizeof(_Float16));

    float* topkVals = out + 4 * (size_t)T;        // last output in the tuple

    hnm_normalize_kernel<<<N, DIM, 0, stream>>>(memory, memH);
    hnm_mine_kernel<<<B / 16, 256, 0, stream>>>(memH, src, dst, labels,
                                                topkVals, topkIdx, N);
    hnm_assemble_kernel<<<1, 256, 0, stream>>>(src, dst, labels, ts,
                                               topkIdx, out, B, N);
    (void)n_in; (void)out_size; (void)ws_size;
}
